// FusionBlock_88149908783839
// MI455X (gfx1250) — matrix operand order repaired and checked by reading, not compiled, not run on hardware
//
#include <hip/hip_runtime.h>
#include <hip/hip_bf16.h>

// ---------------------------------------------------------------------------
// FusionBlock for MI455X (gfx1250): all large GEMMs via v_wmma_f32_16x16x32_f16,
// tile staging via global_load_async_to_lds_b128 (ASYNCcnt) with double buffering.
// ---------------------------------------------------------------------------

typedef _Float16 v16h __attribute__((ext_vector_type(16)));
typedef _Float16 v8h  __attribute__((ext_vector_type(8)));
typedef float    v8f  __attribute__((ext_vector_type(8)));

#define NV   4
#define BB   4
#define NN   256
#define CC   768
#define HH   12
#define HD   64
#define HID  3072
#define C3   2304
#define MROWS (BB*NN)          // 1024 rows per view

__device__ __forceinline__ v8f wmma16(v16h a, v16h b, v8f c) {
  return __builtin_amdgcn_wmma_f32_16x16x32_f16(false, a, false, b, (short)0, c,
                                                false, false);
}

// Low 32 bits of a generic pointer into LDS == LDS byte offset.
template <typename T>
__device__ __forceinline__ unsigned lds_off(T* p) {
  return (unsigned)(unsigned long long)p;
}

// Async DMA: 16 bytes per lane, global -> LDS, tracked by ASYNCcnt.
__device__ __forceinline__ void async_b128(unsigned lds, const void* gaddr) {
  asm volatile("global_load_async_to_lds_b128 %0, %1, off"
               :: "v"(lds), "v"(gaddr) : "memory");
}
__device__ __forceinline__ void wait_async0() {
  asm volatile("s_wait_asynccnt 0" ::: "memory");
}

// A fragment (16x32 f16): lane l -> row M = l&15; 16 contiguous K halfs at
// offset 16*(l>=16).
__device__ __forceinline__ v16h frag_a(const _Float16* __restrict__ base,
                                       int rowStride, int lane) {
  const _Float16* p = base + (lane & 15) * rowStride + ((lane >> 4) << 4);
  v8h lo = *(const v8h*)(p);
  v8h hi = *(const v8h*)(p + 8);
  return __builtin_shufflevector(lo, hi, 0,1,2,3,4,5,6,7,8,9,10,11,12,13,14,15);
}

// B fragment (32x16 f16): lane l -> col N = l&15; 16 contiguous K halfs at
// offset 16*(l>=16).  Source rows are the N dimension (NT gemm: W[n][k]).
__device__ __forceinline__ v16h frag_b(const _Float16* __restrict__ base,
                                       int rowStride, int lane) {
  const _Float16* p = base + (lane & 15) * rowStride + ((lane >> 4) << 4);
  v8h lo = *(const v8h*)(p);
  v8h hi = *(const v8h*)(p + 8);
  return __builtin_shufflevector(lo, hi, 0,1,2,3,4,5,6,7,8,9,10,11,12,13,14,15);
}

// B fragment gathered from row-major V (rows = K, cols = N): transpose gather.
__device__ __forceinline__ v16h frag_bT(const _Float16* __restrict__ sV,
                                        int rowStride, int k0, int n0, int lane) {
  int d = n0 + (lane & 15);
  int kb = k0 + ((lane >> 4) << 4);
  v16h r;
#pragma unroll
  for (int t = 0; t < 16; ++t) r[t] = sV[(kb + t) * rowStride + d];
  return r;
}

// ---------------------------------------------------------------------------
// fp32 -> fp16 weight conversion
// ---------------------------------------------------------------------------
__global__ void cvt_kernel(const float* __restrict__ in,
                           _Float16* __restrict__ out, long n) {
  long i = (long)blockIdx.x * blockDim.x + threadIdx.x;
  long stride = (long)gridDim.x * blockDim.x;
  for (; i < n; i += stride) out[i] = (_Float16)in[i];
}

// ---------------------------------------------------------------------------
// LayerNorm 1: X[(b*NV+v)][n][c] fp32 -> Xn[v][b*N+n][c] f16 (shared g/b)
// ---------------------------------------------------------------------------
__global__ void __launch_bounds__(256)
ln1_kernel(const float* __restrict__ X, const float* __restrict__ g,
           const float* __restrict__ bta, _Float16* __restrict__ out) {
  __shared__ float red[256];
  int t = blockIdx.x;                    // v*1024 + b*256 + n
  int v = t >> 10, bn = t & 1023, b = bn >> 8, n = bn & 255;
  const float* x = X + ((long)(b * NV + v) * NN + n) * CC;
  int tid = threadIdx.x;
  float s = 0.f, ss = 0.f;
  for (int c = tid; c < CC; c += 256) { float q = x[c]; s += q; ss += q * q; }
  red[tid] = s; __syncthreads();
  for (int o = 128; o > 0; o >>= 1) { if (tid < o) red[tid] += red[tid + o]; __syncthreads(); }
  float mean = red[0] * (1.f / CC); __syncthreads();
  red[tid] = ss; __syncthreads();
  for (int o = 128; o > 0; o >>= 1) { if (tid < o) red[tid] += red[tid + o]; __syncthreads(); }
  float var = red[0] * (1.f / CC) - mean * mean;
  float rstd = rsqrtf(var + 1e-6f);
  _Float16* o16 = out + (long)t * CC;
  for (int c = tid; c < CC; c += 256)
    o16[c] = (_Float16)((x[c] - mean) * rstd * g[c] + bta[c]);
}

// LayerNorm 2: per-view gamma/beta, fp32 in -> f16 out (same [v][m][c] layout)
__global__ void __launch_bounds__(256)
ln2_kernel(const float* __restrict__ X, const float* __restrict__ g,
           const float* __restrict__ bta, _Float16* __restrict__ out) {
  __shared__ float red[256];
  int t = blockIdx.x;                    // v*1024 + m
  int v = t >> 10;
  const float* x = X + (long)t * CC;
  const float* gv = g + v * CC; const float* bv = bta + v * CC;
  int tid = threadIdx.x;
  float s = 0.f, ss = 0.f;
  for (int c = tid; c < CC; c += 256) { float q = x[c]; s += q; ss += q * q; }
  red[tid] = s; __syncthreads();
  for (int o = 128; o > 0; o >>= 1) { if (tid < o) red[tid] += red[tid + o]; __syncthreads(); }
  float mean = red[0] * (1.f / CC); __syncthreads();
  red[tid] = ss; __syncthreads();
  for (int o = 128; o > 0; o >>= 1) { if (tid < o) red[tid] += red[tid + o]; __syncthreads(); }
  float var = red[0] * (1.f / CC) - mean * mean;
  float rstd = rsqrtf(var + 1e-6f);
  _Float16* o16 = out + (long)t * CC;
  for (int c = tid; c < CC; c += 256)
    o16[c] = (_Float16)((x[c] - mean) * rstd * gv[c] + bv[c]);
}

// ---------------------------------------------------------------------------
// Generic batched NT WMMA GEMM: out[m][n] = sum_k A[m][k]*Bw[n][k] (+epilogue)
// Block = 256 thr (8 waves), tile 64x64, BK=32, double-buffered async-to-LDS
// staging.  blockIdx.z = view.
// EPI: 0 f16 store, 1 bias+f16, 2 2*(x+bias)->f32, 3 bias+gelu->f16,
//      4 bias+resid->f32 and f16
// ---------------------------------------------------------------------------
template <int EPI>
__global__ void __launch_bounds__(256)
gemm_nt(const _Float16* __restrict__ A, const _Float16* __restrict__ Bw,
        const float* __restrict__ bias, float* __restrict__ outF,
        _Float16* __restrict__ outH, const float* __restrict__ resid,
        int M, int N, int K, long sA, long sB, long sBias, long sOut) {
  __shared__ _Float16 sAt[2][64 * 40];
  __shared__ _Float16 sBt[2][64 * 40];
  const int v = blockIdx.z;
  A += (long)v * sA; Bw += (long)v * sB;
  if (EPI != 0) bias += (long)v * sBias;
  if (EPI == 2 || EPI == 4) outF += (long)v * sOut;
  if (EPI != 2) outH += (long)v * sOut;
  if (EPI == 4) resid += (long)v * sOut;

  const int m0 = blockIdx.y * 64, n0 = blockIdx.x * 64;
  const int tid = threadIdx.x, lane = tid & 31, wv = tid >> 5;
  const int mt = wv >> 1;               // 0..3
  const int ntb = (wv & 1) * 2;         // 0 or 2 (owns tiles ntb, ntb+1)
  v8f acc0 = {}, acc1 = {};

  const int lr = tid >> 2;              // 0..63
  const int lc = (tid & 3) << 3;        // 0,8,16,24
  const _Float16* gA = A + (long)(m0 + lr) * K + lc;   // + k0 per step
  const _Float16* gB = Bw + (long)(n0 + lr) * K + lc;
  const unsigned ldA0 = lds_off(&sAt[0][lr * 40 + lc]);
  const unsigned ldB0 = lds_off(&sBt[0][lr * 40 + lc]);
  const unsigned ldA1 = lds_off(&sAt[1][lr * 40 + lc]);
  const unsigned ldB1 = lds_off(&sBt[1][lr * 40 + lc]);

  // prologue: async DMA first K-slice into buffer 0
  async_b128(ldA0, gA);
  async_b128(ldB0, gB);
  wait_async0();
  __syncthreads();

  const int nk = K >> 5;
  for (int ks = 0; ks < nk; ++ks) {
    const int cur = ks & 1;
    if (ks + 1 < nk) {                  // prefetch next slice into other buffer
      async_b128(cur ? ldA0 : ldA1, gA + (ks + 1) * 32);
      async_b128(cur ? ldB0 : ldB1, gB + (ks + 1) * 32);
    }
    v16h af  = frag_a(&sAt[cur][mt * 16 * 40], 40, lane);
    v16h bf0 = frag_b(&sBt[cur][ntb * 16 * 40], 40, lane);
    v16h bf1 = frag_b(&sBt[cur][(ntb + 1) * 16 * 40], 40, lane);
    acc0 = wmma16(af, bf0, acc0);
    acc1 = wmma16(af, bf1, acc1);
    wait_async0();                      // next buffer landed (overlapped w/ WMMA)
    __syncthreads();
  }

  const int gm0 = m0 + mt * 16 + ((lane >> 4) << 3);
#pragma unroll
  for (int tt = 0; tt < 2; ++tt) {
    v8f acc = tt ? acc1 : acc0;
    int gn = n0 + (ntb + tt) * 16 + (lane & 15);
    float bz = (EPI != 0) ? bias[gn] : 0.f;
#pragma unroll
    for (int r = 0; r < 8; ++r) {
      long idx = (long)(gm0 + r) * N + gn;
      float x = acc[r];
      if (EPI == 0) outH[idx] = (_Float16)x;
      else if (EPI == 1) outH[idx] = (_Float16)(x + bz);
      else if (EPI == 2) outF[idx] = 2.f * (x + bz);
      else if (EPI == 3) {
        float t = x + bz;
        outH[idx] = (_Float16)(0.5f * t * (1.f + erff(t * 0.70710678118654752f)));
      } else {
        float t = x + bz + resid[idx];
        outF[idx] = t; outH[idx] = (_Float16)t;
      }
    }
  }
}

// ---------------------------------------------------------------------------
// Fused 4x4 pairwise cross-view attention, mean over j folded into softmax.
// Grid: (band 0..3, head 0..11, iv*B+b).  Dynamic LDS; async K/V staging.
// QKV layout: [v][b*N+n][3C], q/k/v at col s*768 + h*64.
// ---------------------------------------------------------------------------
#define QSTR 72
#define KVSTR 72
#define SSTR 264
#define PSTR 264

__global__ void __launch_bounds__(256)
fusion_attn_kernel(const _Float16* __restrict__ QKV, _Float16* __restrict__ ctx) {
  extern __shared__ char smem[];
  _Float16* sQ  = (_Float16*)smem;                       // 64 x 72
  _Float16* sKV = sQ + 64 * QSTR;                        // 256 x 72
  float*    sS  = (float*)(sKV + 256 * KVSTR);           // 64 x 264 f32
  _Float16* sP  = (_Float16*)(sS + 64 * SSTR);           // 64 x 264 f16

  const int band = blockIdx.x, h = blockIdx.y;
  const int vb = blockIdx.z, iv = vb >> 2, b = vb & 3;
  const int tid = threadIdx.x, lane = tid & 31, wv = tid >> 5;
  const long rowBase = (long)iv * MROWS + b * NN;

  // async-load Q band (64 x 64); waited on by first wait_async0 below
  {
    int r = tid >> 2, c = (tid & 3) << 4;
    const _Float16* src = QKV + (rowBase + band * 64 + r) * (long)C3 + h * HD + c;
    async_b128(lds_off(&sQ[r * QSTR + c]), src);
    async_b128(lds_off(&sQ[r * QSTR + c + 8]), src + 8);
  }

  v8f o0 = {}, o1 = {};
  const int omt = wv >> 1;            // ctx M tile
  const int odt = (wv & 1) * 2;       // ctx D tiles odt, odt+1

  for (int j = 0; j < NV; ++j) {
    __syncthreads();
    // K_j -> sKV (256 x 64) via async DMA
    {
      const _Float16* src = QKV + ((long)j * MROWS + b * NN + tid) * C3 + CC + h * HD;
#pragma unroll
      for (int c = 0; c < HD; c += 8)
        async_b128(lds_off(&sKV[tid * KVSTR + c]), src + c);
    }
    wait_async0();
    __syncthreads();
    // S = Q K^T * scale  (M=64, N=256, K=64)
    {
      const int smt = wv >> 1, sntb = (wv & 1) * 8;
#pragma unroll 1
      for (int nt = 0; nt < 8; ++nt) {
        v8f s = {};
#pragma unroll
        for (int kk = 0; kk < HD; kk += 32) {
          v16h af = frag_a(&sQ[smt * 16 * QSTR + kk], QSTR, lane);
          v16h bf = frag_b(&sKV[(sntb + nt) * 16 * KVSTR + kk], KVSTR, lane);
          s = wmma16(af, bf, s);
        }
        int gn = (sntb + nt) * 16 + (lane & 15);
        int gm = smt * 16 + ((lane >> 4) << 3);
#pragma unroll
        for (int r = 0; r < 8; ++r) sS[(gm + r) * SSTR + gn] = s[r] * 0.125f;
      }
    }
    __syncthreads();
    // row softmax (threads 0..63 own one row each); fold 1/NV mean into P
    if (tid < 64) {
      float* row = &sS[tid * SSTR];
      float mx = -1e30f;
      for (int n = 0; n < NN; ++n) mx = fmaxf(mx, row[n]);
      float sum = 0.f;
      for (int n = 0; n < NN; ++n) { float e = __expf(row[n] - mx); row[n] = e; sum += e; }
      float inv = 0.25f / sum;
      _Float16* pr = &sP[tid * PSTR];
      for (int n = 0; n < NN; ++n) pr[n] = (_Float16)(row[n] * inv);
    }
    __syncthreads();
    // V_j -> sKV (overwrite K) via async DMA
    {
      const _Float16* src = QKV + ((long)j * MROWS + b * NN + tid) * C3 + 2 * CC + h * HD;
#pragma unroll
      for (int c = 0; c < HD; c += 8)
        async_b128(lds_off(&sKV[tid * KVSTR + c]), src + c);
    }
    wait_async0();
    __syncthreads();
    // O += P @ V  (M=64, N=64, K=256), B fragment gathered (transpose) from V
#pragma unroll 1
    for (int kk = 0; kk < NN; kk += 32) {
      v16h af = frag_a(&sP[omt * 16 * PSTR + kk], PSTR, lane);
      v16h b0 = frag_bT(sKV, KVSTR, kk, odt * 16, lane);
      v16h b1 = frag_bT(sKV, KVSTR, kk, (odt + 1) * 16, lane);
      o0 = wmma16(af, b0, o0);
      o1 = wmma16(af, b1, o1);
    }
  }
  // store ctx_mean f16 [v][m][C] at head column block
  {
    int dcol = lane & 15;
    int gm0 = omt * 16 + ((lane >> 4) << 3);
#pragma unroll
    for (int r = 0; r < 8; ++r) {
      long row = rowBase + band * 64 + gm0 + r;
      ctx[row * CC + h * HD + odt * 16 + dcol]       = (_Float16)o0[r];
      ctx[row * CC + h * HD + (odt + 1) * 16 + dcol] = (_Float16)o1[r];
    }
  }
}

// ---------------------------------------------------------------------------
// qh[v][d] = sum_c query[c] * wq[v][d][c] + bq   (wq = rows 0..767 of mha_in_w)
// ---------------------------------------------------------------------------
__global__ void qh_kernel(const float* __restrict__ q0, const float* __restrict__ w,
                          const float* __restrict__ bias, float* __restrict__ qh) {
  int idx = blockIdx.x * 256 + threadIdx.x;    // < NV*CC
  int v = idx / CC, d = idx % CC;
  const float* wr = w + ((long)v * C3 + d) * CC;
  float s = bias[v * C3 + d];
  for (int c = 0; c < CC; ++c) s += q0[c] * wr[c];
  qh[idx] = s;
}

// ---------------------------------------------------------------------------
// Learnable-query attention per (v,b): softmax over N=256 keys per head.
// ---------------------------------------------------------------------------
__global__ void __launch_bounds__(256)
xattn_kernel(const float* __restrict__ qh, const _Float16* __restrict__ kh,
             const _Float16* __restrict__ vh, float* __restrict__ c2) {
  __shared__ float red[256];
  __shared__ float sa[256];
  int vb = blockIdx.x, v = vb >> 2, b = vb & 3;
  int tid = threadIdx.x;
  const long rb = (long)v * MROWS + b * NN;
  for (int h = 0; h < HH; ++h) {
    const _Float16* krow = kh + (rb + tid) * CC + h * HD;
    const float* qv = qh + v * CC + h * HD;
    float dot = 0.f;
#pragma unroll
    for (int e = 0; e < HD; ++e) dot += qv[e] * (float)krow[e];
    dot *= 0.125f;
    red[tid] = dot; __syncthreads();
    for (int o = 128; o > 0; o >>= 1) { if (tid < o) red[tid] = fmaxf(red[tid], red[tid + o]); __syncthreads(); }
    float mx = red[0]; __syncthreads();
    float ee = __expf(dot - mx);
    red[tid] = ee; __syncthreads();
    for (int o = 128; o > 0; o >>= 1) { if (tid < o) red[tid] += red[tid + o]; __syncthreads(); }
    float inv = 1.f / red[0];
    sa[tid] = ee * inv; __syncthreads();
    if (tid < HD) {
      float acc = 0.f;
      for (int n = 0; n < NN; ++n)
        acc += sa[n] * (float)vh[(rb + n) * CC + h * HD + tid];
      c2[(long)vb * CC + h * HD + tid] = acc;
    }
    __syncthreads();
  }
}

// ---------------------------------------------------------------------------
// Final projection + broadcast to all N positions:
// out[b][n][v*C+d] = sum_c c2[v][b][c]*mha_out_w[v][d][c] + mha_out_b[v][d]
// ---------------------------------------------------------------------------
__global__ void out_kernel(const float* __restrict__ c2, const float* __restrict__ w,
                           const float* __restrict__ bias, float* __restrict__ out) {
  int idx = blockIdx.x * 256 + threadIdx.x;    // < NV*BB*CC
  int vb = idx / CC, d = idx % CC, v = vb >> 2, b = vb & 3;
  const float* wr = w + ((long)v * CC + d) * CC;
  const float* cc = c2 + (long)vb * CC;
  float s = bias[v * CC + d];
  for (int c = 0; c < CC; ++c) s += cc[c] * wr[c];
  for (int n = 0; n < NN; ++n)
    out[((long)b * NN + n) * (NV * CC) + v * CC + d] = s;
}

// ---------------------------------------------------------------------------
// Host launcher
// ---------------------------------------------------------------------------
extern "C" void kernel_launch(void* const* d_in, const int* in_sizes, int n_in,
                              void* d_out, int out_size, void* d_ws, size_t ws_size,
                              hipStream_t stream) {
  const float* X        = (const float*)d_in[0];
  const float* norm1_g  = (const float*)d_in[1];
  const float* norm1_b  = (const float*)d_in[2];
  const float* qkv_w    = (const float*)d_in[3];
  const float* proj_w   = (const float*)d_in[4];
  const float* proj_b   = (const float*)d_in[5];
  const float* norm2_g  = (const float*)d_in[6];
  const float* norm2_b  = (const float*)d_in[7];
  const float* fc1_w    = (const float*)d_in[8];
  const float* fc1_b    = (const float*)d_in[9];
  const float* fc2_w    = (const float*)d_in[10];
  const float* fc2_b    = (const float*)d_in[11];
  const float* query    = (const float*)d_in[12];
  const float* mha_in_w = (const float*)d_in[13];
  const float* mha_in_b = (const float*)d_in[14];
  const float* mha_out_w= (const float*)d_in[15];
  const float* mha_out_b= (const float*)d_in[16];
  float* out = (float*)d_out;

  // ---- deterministic workspace carving ----
  char* ws = (char*)d_ws;
  auto carve = [&](size_t bytes) -> char* {
    char* p = ws; ws += (bytes + 255) & ~(size_t)255; return p;
  };
  const long nQKVW = (long)NV * C3 * CC;
  const long nPROJ = (long)NV * CC * CC;
  const long nFC   = (long)NV * HID * CC;
  const long nACT  = (long)NV * MROWS * CC;     // [v][1024][768]
  const long nQKVA = (long)NV * MROWS * C3;
  const long nH1   = (long)NV * MROWS * HID;

  _Float16* qkvw_h  = (_Float16*)carve(nQKVW * 2);
  _Float16* projw_h = (_Float16*)carve(nPROJ * 2);
  _Float16* fc1w_h  = (_Float16*)carve(nFC * 2);
  _Float16* fc2w_h  = (_Float16*)carve(nFC * 2);
  _Float16* mhaw_h  = (_Float16*)carve(nQKVW * 2);
  _Float16* Xn      = (_Float16*)carve(nACT * 2);   // LN1 out; reused as LN2 out
  _Float16* QKVa    = (_Float16*)carve(nQKVA * 2);  // qkv acts; reused for kh/vh
  _Float16* CTXm    = (_Float16*)carve(nACT * 2);
  float*    X2      = (float*)   carve(nACT * 4);
  _Float16* H1      = (_Float16*)carve(nH1 * 2);
  float*    Xf      = (float*)   carve(nACT * 4);
  _Float16* Xf_h    = (_Float16*)carve(nACT * 2);
  float*    QH      = (float*)   carve((long)NV * CC * 4);
  float*    C2      = (float*)   carve((long)NV * BB * CC * 4);
  _Float16* Hn = Xn;                       // alias (Xn dead after QKV gemm)
  _Float16* KH = QKVa;                     // alias (QKVa dead after attention)
  _Float16* VH = QKVa + nACT;

  // ---- 1. weight conversion ----
  cvt_kernel<<<4096, 256, 0, stream>>>(qkv_w,  qkvw_h,  nQKVW);
  cvt_kernel<<<2048, 256, 0, stream>>>(proj_w, projw_h, nPROJ);
  cvt_kernel<<<4096, 256, 0, stream>>>(fc1_w,  fc1w_h,  nFC);
  cvt_kernel<<<4096, 256, 0, stream>>>(fc2_w,  fc2w_h,  nFC);
  cvt_kernel<<<4096, 256, 0, stream>>>(mha_in_w, mhaw_h, nQKVW);

  // ---- 2. LN1 ----
  ln1_kernel<<<NV * MROWS, 256, 0, stream>>>(X, norm1_g, norm1_b, Xn);

  // ---- 3. QKV gemm: [1024x768] x [2304x768]^T per view, f16 out ----
  gemm_nt<0><<<dim3(C3 / 64, MROWS / 64, NV), 256, 0, stream>>>(
      Xn, qkvw_h, nullptr, nullptr, QKVa, nullptr,
      MROWS, C3, CC, (long)MROWS * CC, (long)C3 * CC, 0, (long)MROWS * C3);

  // ---- 4. fused pairwise attention -> ctx_mean f16 ----
  {
    size_t smem = 64 * QSTR * 2 + 256 * KVSTR * 2 + 64 * SSTR * 4 + 64 * PSTR * 2;
    fusion_attn_kernel<<<dim3(4, HH, NV * BB), 256, smem, stream>>>(QKVa, CTXm);
  }

  // ---- 5. proj: 2*(ctx @ proj_w^T + b) -> X2 f32 ----
  gemm_nt<2><<<dim3(CC / 64, MROWS / 64, NV), 256, 0, stream>>>(
      CTXm, projw_h, proj_b, X2, nullptr, nullptr,
      MROWS, CC, CC, (long)MROWS * CC, (long)CC * CC, CC, (long)MROWS * CC);

  // ---- 6. LN2 -> Hn f16 ----
  ln2_kernel<<<NV * MROWS, 256, 0, stream>>>(X2, norm2_g, norm2_b, Hn);

  // ---- 7. fc1 + exact GELU -> H1 f16 ----
  gemm_nt<3><<<dim3(HID / 64, MROWS / 64, NV), 256, 0, stream>>>(
      Hn, fc1w_h, fc1_b, nullptr, H1, nullptr,
      MROWS, HID, CC, (long)MROWS * CC, (long)HID * CC, HID, (long)MROWS * HID);

  // ---- 8. fc2 + bias + residual(X2) -> Xf f32 and Xf_h f16 ----
  gemm_nt<4><<<dim3(CC / 64, MROWS / 64, NV), 256, 0, stream>>>(
      H1, fc2w_h, fc2_b, Xf, Xf_h, X2,
      MROWS, CC, HID, (long)MROWS * HID, (long)CC * HID, CC, (long)MROWS * CC);

  // ---- 9. kh / vh gemms (wk rows 768.., wv rows 1536.. of mha_in_w) ----
  gemm_nt<1><<<dim3(CC / 64, MROWS / 64, NV), 256, 0, stream>>>(
      Xf_h, mhaw_h + (long)CC * CC, mha_in_b + CC, nullptr, KH, nullptr,
      MROWS, CC, CC, (long)MROWS * CC, (long)C3 * CC, C3, (long)MROWS * CC);
  gemm_nt<1><<<dim3(CC / 64, MROWS / 64, NV), 256, 0, stream>>>(
      Xf_h, mhaw_h + (long)2 * CC * CC, mha_in_b + 2 * CC, nullptr, VH, nullptr,
      MROWS, CC, CC, (long)MROWS * CC, (long)C3 * CC, C3, (long)MROWS * CC);

  // ---- 10. qh (tiny, fp32 VALU) ----
  qh_kernel<<<(NV * CC) / 256, 256, 0, stream>>>(query, mha_in_w, mha_in_b, QH);

  // ---- 11. learnable-query attention -> c2 ----
  xattn_kernel<<<NV * BB, 256, 0, stream>>>(QH, KH, VH, C2);

  // ---- 12. final projection + broadcast over N ----
  out_kernel<<<(NV * BB * CC) / 256, 256, 0, stream>>>(C2, mha_out_w, mha_out_b, out);
}
